// SwinBlock_25271587570169
// MI455X (gfx1250) — matrix operand order repaired and checked by reading, not compiled, not run on hardware
//
#include <hip/hip_runtime.h>
#include <hip/hip_bf16.h>
#include <math.h>

#define BATCH 32
#define IMH 56
#define IMW 56
#define CC 192
#define HEADS 6
#define HD 32
#define WS 7
#define SSH 3
#define NTOK 49
#define NWIN 64            // (56/7)*(56/7)
#define BWIN (BATCH*NWIN)  // 2048
#define TOKENS (BATCH*IMH*IMW)  // 100352
#define HID 768
#define EPSLN 1e-5f

typedef __attribute__((ext_vector_type(16))) _Float16 v16h;
typedef __attribute__((ext_vector_type(8)))  float    v8f;
typedef _Float16 f16;
typedef unsigned int u32;

// ---------------- fragment loaders (contiguous k order, same for A and B) ----------------

// A matrix 16x32 (MxK), row-major with leading dim ld.
// lane<16: row M=lane, K = 0..15 (elems 0..15)
// lane>=16: row M=lane-16, K = 16..31
__device__ inline v16h load_a(const f16* base, int ld, int lane) {
  int m = lane & 15;
  int koff = (lane < 16) ? 0 : 16;
  const f16* r = base + m * ld + koff;
  union { v16h v; u32 u[8]; } t;
#pragma unroll
  for (int p = 0; p < 4; ++p) t.u[p] = *(const u32*)(r + 2 * p);
#pragma unroll
  for (int p = 0; p < 4; ++p) t.u[4 + p] = *(const u32*)(r + 8 + 2 * p);
  return t.v;
}

// B matrix 32x16 (KxN) supplied TRANSPOSED: bt[n][k], leading dim ld (>=32).
// lane<16: col N=lane, K=0..15 ; lane>=16: col N=lane-16, K=16..31
__device__ inline v16h load_b_t(const f16* bt, int ld, int lane) {
  int n = lane & 15;
  int kb = (lane < 16) ? 0 : 16;
  const f16* r = bt + n * ld + kb;
  union { v16h v; u32 u[8]; } t;
#pragma unroll
  for (int p = 0; p < 8; ++p) t.u[p] = *(const u32*)(r + 2 * p);
  return t.v;
}

// B matrix 32x16 (KxN) row-major b[k][n] (strided reads; used from LDS)
__device__ inline v16h load_b_rm(const f16* b, int ld, int lane) {
  int n = lane & 15;
  int kb = (lane < 16) ? 0 : 16;
  union { v16h v; f16 h[16]; } t;
#pragma unroll
  for (int e = 0; e < 16; ++e) t.h[e] = b[(kb + e) * ld + n];
  return t.v;
}

__device__ inline v8f wmma_f16(v16h a, v16h b, v8f c) {
  return __builtin_amdgcn_wmma_f32_16x16x32_f16(false, a, false, b, (short)0, c,
                                                false, false);
}

// 16x64 output strip: one A fragment per K-step feeds 4 WMMAs (4 N-tiles).
// A tile row-major [16 x K] (lda=K), Bt = 4 consecutive [16 x K] column tiles.
__device__ inline void gemm16x64(const f16* A, const f16* Bt, int Kdim, int lane,
                                 v8f acc[4]) {
  for (int kk = 0; kk < Kdim; kk += 32) {
    v16h a = load_a(A + kk, Kdim, lane);
#pragma unroll
    for (int q = 0; q < 4; ++q) {
      v16h b = load_b_t(Bt + (size_t)q * 16 * Kdim + kk, Kdim, lane);
      acc[q] = wmma_f16(a, b, acc[q]);
    }
  }
}

__device__ inline float wave_sum(float v) {
#pragma unroll
  for (int off = 16; off > 0; off >>= 1) v += __shfl_xor(v, off, 32);
  return v;
}

// window token -> image pixel (same mapping for gather with roll(-3) and
// scatter with roll(+3): the residual stays aligned)
__device__ inline int tok2pix(int t) {
  int win = t / NTOK, n = t - win * NTOK;
  int b = win / NWIN, wb = win - b * NWIN;
  int wh = wb >> 3, ww = wb & 7;
  int i = n / WS, j = n - i * WS;
  int h = (wh * WS + i + SSH) % IMH;
  int w = (ww * WS + j + SSH) % IMW;
  return (b * IMH + h) * IMW + w;
}

// ---------------- kernels ----------------

// fp32 W[K][N] -> f16 Wt[N][K]
__global__ void k_wcvt(const float* __restrict__ w, f16* __restrict__ wt,
                       int K, int N) {
  int i = blockIdx.x * blockDim.x + threadIdx.x;
  if (i < K * N) {
    int n = i / K, k = i - n * K;
    wt[i] = (f16)w[(size_t)k * N + n];
  }
}

// LN1 + roll(-3,-3) + window partition, one wave per window-token
__global__ void k_ln1(const float* __restrict__ x, const float* __restrict__ g,
                      const float* __restrict__ bb, f16* __restrict__ xw) {
  int wid = (blockIdx.x * blockDim.x + threadIdx.x) >> 5;
  int lane = threadIdx.x & 31;
  int pix = tok2pix(wid);
  const float* px = x + (size_t)pix * CC;
  float v[6], s = 0.f, s2 = 0.f;
#pragma unroll
  for (int e = 0; e < 6; ++e) {
    float t = px[lane + 32 * e];
    v[e] = t; s += t; s2 += t * t;
  }
  s = wave_sum(s); s2 = wave_sum(s2);
  float mu = s * (1.0f / CC);
  float inv = rsqrtf(s2 * (1.0f / CC) - mu * mu + EPSLN);
  f16* out = xw + (size_t)wid * CC;
#pragma unroll
  for (int e = 0; e < 6; ++e) {
    int c = lane + 32 * e;
    out[c] = (f16)((v[e] - mu) * inv * g[c] + bb[c]);
  }
}

// LN2 on y (pixel order), one wave per token
__global__ void k_ln2(const float* __restrict__ y, const float* __restrict__ g,
                      const float* __restrict__ bb, f16* __restrict__ m) {
  int wid = (blockIdx.x * blockDim.x + threadIdx.x) >> 5;
  int lane = threadIdx.x & 31;
  const float* py = y + (size_t)wid * CC;
  float v[6], s = 0.f, s2 = 0.f;
#pragma unroll
  for (int e = 0; e < 6; ++e) {
    float t = py[lane + 32 * e];
    v[e] = t; s += t; s2 += t * t;
  }
  s = wave_sum(s); s2 = wave_sum(s2);
  float mu = s * (1.0f / CC);
  float inv = rsqrtf(s2 * (1.0f / CC) - mu * mu + EPSLN);
  f16* out = m + (size_t)wid * CC;
#pragma unroll
  for (int e = 0; e < 6; ++e) {
    int c = lane + 32 * e;
    out[c] = (f16)((v[e] - mu) * inv * g[c] + bb[c]);
  }
}

// QKV GEMM: [100352 x 192] @ [192 x 576]; wave = 16x64 strip.
// Epilogue splits Q(scaled)/K/V  (64 | 192, so each strip hits one of Q/K/V)
__global__ void k_gemm_qkv(const f16* __restrict__ A, const f16* __restrict__ Bt,
                           const float* __restrict__ bias, f16* __restrict__ Qg,
                           f16* __restrict__ Kg, f16* __restrict__ Vg) {
  int wid = blockIdx.x * 8 + (threadIdx.x >> 5);
  int lane = threadIdx.x & 31;
  const int NG = (3 * CC) / 64;  // 9 column groups
  int tg = wid % NG, tm = wid / NG;
  v8f acc[4] = {};
  gemm16x64(A + (size_t)tm * 16 * CC, Bt + (size_t)tg * 64 * CC, CC, lane, acc);
  int n = lane & 15, mb = (lane < 16) ? 0 : 8;
#pragma unroll
  for (int q = 0; q < 4; ++q) {
    int col = tg * 64 + q * 16 + n;
    float bv = bias[col];
    int sel = col / CC, cc = col - sel * CC;
    f16* dst = (sel == 0) ? Qg : ((sel == 1) ? Kg : Vg);
    float mul = (sel == 0) ? 0.17677669529663687f : 1.0f;  // hd^-0.5 on Q
#pragma unroll
    for (int r = 0; r < 8; ++r) {
      int m = tm * 16 + mb + r;
      dst[(size_t)m * CC + cc] = (f16)((acc[q][r] + bv) * mul);
    }
  }
}

// attention: one block (4 waves) per (window, head)
__global__ void k_attn(const f16* __restrict__ Qg, const f16* __restrict__ Kg,
                       const f16* __restrict__ Vg,
                       const float* __restrict__ bias_table,
                       f16* __restrict__ AO) {
  __shared__ f16 sQ[64 * 32];
  __shared__ f16 sK[64 * 32];
  __shared__ f16 sV[64 * 32];
  __shared__ f16 sP[64 * 64];
  __shared__ float sS[64 * 64];

  int whid = blockIdx.x;
  int win = whid / HEADS;
  int head = whid - win * HEADS;
  int tid = threadIdx.x;
  int lane = tid & 31, wave = tid >> 5;

  // stage Q/K/V head-slices (49x32, zero pad to 64 rows)
  for (int u = tid; u < 1024; u += 128) {
    int r = u >> 4, c2 = (u & 15) << 1;
    u32 q = 0, k = 0, v = 0;
    if (r < NTOK) {
      size_t base = (size_t)(win * NTOK + r) * CC + head * HD + c2;
      q = *(const u32*)(Qg + base);
      k = *(const u32*)(Kg + base);
      v = *(const u32*)(Vg + base);
    }
    *(u32*)&sQ[r * 32 + c2] = q;
    *(u32*)&sK[r * 32 + c2] = k;
    *(u32*)&sV[r * 32 + c2] = v;
  }
  __syncthreads();

  // S = Q * K^T  (each wave: 16 query rows x 64 key cols)
  int m0 = wave * 16;
  {
    v16h a = load_a(sQ + m0 * 32, 32, lane);
#pragma unroll
    for (int nt = 0; nt < 4; ++nt) {
      v16h b = load_b_t(sK + (nt * 16) * 32, 32, lane);  // K rows are B columns
      v8f c = {};
      c = wmma_f16(a, b, c);
      int n = lane & 15, mb = (lane < 16) ? 0 : 8;
#pragma unroll
      for (int r = 0; r < 8; ++r) sS[(m0 + mb + r) * 64 + nt * 16 + n] = c[r];
    }
  }
  __syncthreads();

  // bias + shift-mask + softmax (f32), write P as f16 (padded cols/rows = 0)
  if (tid < 64) {
    int n = tid;
    if (n < NTOK) {
      int wb = win % NWIN, wh = wb >> 3, ww = wb & 7;
      int i1 = n / WS, j1 = n - i1 * WS;
      int gh1 = wh * WS + i1, gw1 = ww * WS + j1;
      int reg1 = ((gh1 < 49) ? 0 : (gh1 < 53 ? 1 : 2)) * 3 +
                 ((gw1 < 49) ? 0 : (gw1 < 53 ? 1 : 2));
      float mx = -1e30f;
      for (int m = 0; m < NTOK; ++m) {
        int i2 = m / WS, j2 = m - i2 * WS;
        int rel = (i1 - i2 + WS - 1) * (2 * WS - 1) + (j1 - j2 + WS - 1);
        int gh2 = wh * WS + i2, gw2 = ww * WS + j2;
        int reg2 = ((gh2 < 49) ? 0 : (gh2 < 53 ? 1 : 2)) * 3 +
                   ((gw2 < 49) ? 0 : (gw2 < 53 ? 1 : 2));
        float s = sS[n * 64 + m] + bias_table[rel * HEADS + head] +
                  ((reg1 == reg2) ? 0.0f : -100.0f);
        sS[n * 64 + m] = s;
        mx = fmaxf(mx, s);
      }
      float sum = 0.f;
      for (int m = 0; m < NTOK; ++m) {
        float e = expf(sS[n * 64 + m] - mx);
        sS[n * 64 + m] = e;
        sum += e;
      }
      float inv = 1.0f / sum;
      for (int m = 0; m < NTOK; ++m) sP[n * 64 + m] = (f16)(sS[n * 64 + m] * inv);
      for (int m = NTOK; m < 64; ++m) sP[n * 64 + m] = (f16)0.0f;
    } else {
      for (int m = 0; m < 64; ++m) sP[n * 64 + m] = (f16)0.0f;
    }
  }
  __syncthreads();

  // O = P * V  (each wave: 16 rows x 32 hd cols, K = 64 keys)
  {
    v16h a0 = load_a(sP + m0 * 64, 64, lane);       // keys 0..31
    v16h a1 = load_a(sP + m0 * 64 + 32, 64, lane);  // keys 32..63
#pragma unroll
    for (int nt = 0; nt < 2; ++nt) {
      v8f c = {};
      c = wmma_f16(a0, load_b_rm(sV + nt * 16, 32, lane), c);
      c = wmma_f16(a1, load_b_rm(sV + 32 * 32 + nt * 16, 32, lane), c);
      int n = lane & 15, mb = (lane < 16) ? 0 : 8;
#pragma unroll
      for (int r = 0; r < 8; ++r) {
        int m = m0 + mb + r;
        if (m < NTOK)
          AO[(size_t)(win * NTOK + m) * CC + head * HD + nt * 16 + n] = (f16)c[r];
      }
    }
  }
}

// proj GEMM + reverse/roll(+3) scatter + residual -> y (f32, pixel order)
__global__ void k_gemm_proj(const f16* __restrict__ A, const f16* __restrict__ Bt,
                            const float* __restrict__ bias,
                            const float* __restrict__ x, float* __restrict__ y) {
  int wid = blockIdx.x * 8 + (threadIdx.x >> 5);
  int lane = threadIdx.x & 31;
  const int NG = CC / 64;  // 3
  int tg = wid % NG, tm = wid / NG;
  v8f acc[4] = {};
  gemm16x64(A + (size_t)tm * 16 * CC, Bt + (size_t)tg * 64 * CC, CC, lane, acc);
  int n = lane & 15, mb = (lane < 16) ? 0 : 8;
#pragma unroll
  for (int r = 0; r < 8; ++r) {
    int t = tm * 16 + mb + r;
    size_t prow = (size_t)tok2pix(t) * CC;
#pragma unroll
    for (int q = 0; q < 4; ++q) {
      int col = tg * 64 + q * 16 + n;
      y[prow + col] = x[prow + col] + acc[q][r] + bias[col];
    }
  }
}

// FC1 GEMM + exact GELU -> f16
__global__ void k_gemm_fc1(const f16* __restrict__ A, const f16* __restrict__ Bt,
                           const float* __restrict__ bias, f16* __restrict__ h1) {
  int wid = blockIdx.x * 8 + (threadIdx.x >> 5);
  int lane = threadIdx.x & 31;
  const int NG = HID / 64;  // 12
  int tg = wid % NG, tm = wid / NG;
  v8f acc[4] = {};
  gemm16x64(A + (size_t)tm * 16 * CC, Bt + (size_t)tg * 64 * CC, CC, lane, acc);
  int n = lane & 15, mb = (lane < 16) ? 0 : 8;
#pragma unroll
  for (int q = 0; q < 4; ++q) {
    int col = tg * 64 + q * 16 + n;
    float bv = bias[col];
#pragma unroll
    for (int r = 0; r < 8; ++r) {
      int m = tm * 16 + mb + r;
      float v = acc[q][r] + bv;
      float gel = 0.5f * v * (1.0f + erff(v * 0.70710678118654752f));
      h1[(size_t)m * HID + col] = (f16)gel;
    }
  }
}

// FC2 GEMM + bias + residual y -> final f32 output
__global__ void k_gemm_fc2(const f16* __restrict__ A, const f16* __restrict__ Bt,
                           const float* __restrict__ bias,
                           const float* __restrict__ y, float* __restrict__ out) {
  int wid = blockIdx.x * 8 + (threadIdx.x >> 5);
  int lane = threadIdx.x & 31;
  const int NG = CC / 64;  // 3
  int tg = wid % NG, tm = wid / NG;
  v8f acc[4] = {};
  gemm16x64(A + (size_t)tm * 16 * HID, Bt + (size_t)tg * 64 * HID, HID, lane, acc);
  int n = lane & 15, mb = (lane < 16) ? 0 : 8;
#pragma unroll
  for (int r = 0; r < 8; ++r) {
    int m = tm * 16 + mb + r;
    size_t prow = (size_t)m * CC;
#pragma unroll
    for (int q = 0; q < 4; ++q) {
      int col = tg * 64 + q * 16 + n;
      out[prow + col] = y[prow + col] + acc[q][r] + bias[col];
    }
  }
}

// ---------------- launch ----------------

static inline size_t alignup(size_t v) { return (v + 255) & ~(size_t)255; }

extern "C" void kernel_launch(void* const* d_in, const int* in_sizes, int n_in,
                              void* d_out, int out_size, void* d_ws, size_t ws_size,
                              hipStream_t stream) {
  const float* x       = (const float*)d_in[0];
  const float* n1g     = (const float*)d_in[1];
  const float* n1b     = (const float*)d_in[2];
  const float* qkv_w   = (const float*)d_in[3];
  const float* qkv_b   = (const float*)d_in[4];
  const float* proj_w  = (const float*)d_in[5];
  const float* proj_b  = (const float*)d_in[6];
  const float* bias_tb = (const float*)d_in[7];
  const float* n2g     = (const float*)d_in[8];
  const float* n2b     = (const float*)d_in[9];
  const float* fc1_w   = (const float*)d_in[10];
  const float* fc1_b   = (const float*)d_in[11];
  const float* fc2_w   = (const float*)d_in[12];
  const float* fc2_b   = (const float*)d_in[13];

  char* ws = (char*)d_ws;
  size_t off = 0;
  f16* wqkvT = (f16*)(ws + off); off = alignup(off + (size_t)3 * CC * CC * 2);
  f16* wprojT = (f16*)(ws + off); off = alignup(off + (size_t)CC * CC * 2);
  f16* wfc1T = (f16*)(ws + off); off = alignup(off + (size_t)CC * HID * 2);
  f16* wfc2T = (f16*)(ws + off); off = alignup(off + (size_t)HID * CC * 2);
  f16* xw = (f16*)(ws + off); off = alignup(off + (size_t)TOKENS * CC * 2);
  f16* Qb = (f16*)(ws + off); off = alignup(off + (size_t)TOKENS * CC * 2);
  f16* Kb = (f16*)(ws + off); off = alignup(off + (size_t)TOKENS * CC * 2);
  f16* Vb = (f16*)(ws + off); off = alignup(off + (size_t)TOKENS * CC * 2);
  float* yb = (float*)(ws + off); off = alignup(off + (size_t)TOKENS * CC * 4);
  f16* h1 = (f16*)(ws + off); off = alignup(off + (size_t)TOKENS * HID * 2);
  f16* AO = xw;   // reuse: xw consumed by QKV GEMM before attn writes
  f16* mb = Qb;   // reuse: Q consumed by attention before LN2 writes

  // weight convert+transpose
  k_wcvt<<<(CC * 3 * CC + 255) / 256, 256, 0, stream>>>(qkv_w, wqkvT, CC, 3 * CC);
  k_wcvt<<<(CC * CC + 255) / 256, 256, 0, stream>>>(proj_w, wprojT, CC, CC);
  k_wcvt<<<(CC * HID + 255) / 256, 256, 0, stream>>>(fc1_w, wfc1T, CC, HID);
  k_wcvt<<<(HID * CC + 255) / 256, 256, 0, stream>>>(fc2_w, wfc2T, HID, CC);

  // LN1 + shift + partition (100352 tokens, 8 waves/block)
  k_ln1<<<TOKENS / 8, 256, 0, stream>>>(x, n1g, n1b, xw);

  // QKV: (6272 M-tiles) x (9 column groups), 8 waves/block
  k_gemm_qkv<<<(TOKENS / 16) * ((3 * CC) / 64) / 8, 256, 0, stream>>>(
      xw, wqkvT, qkv_b, Qb, Kb, Vb);

  // attention: one block per (window, head)
  k_attn<<<BWIN * HEADS, 128, 0, stream>>>(Qb, Kb, Vb, bias_tb, AO);

  // proj + reverse/roll + residual
  k_gemm_proj<<<(TOKENS / 16) * (CC / 64) / 8, 256, 0, stream>>>(
      AO, wprojT, proj_b, x, yb);

  // LN2
  k_ln2<<<TOKENS / 8, 256, 0, stream>>>(yb, n2g, n2b, mb);

  // FC1 + GELU
  k_gemm_fc1<<<(TOKENS / 16) * (HID / 64) / 8, 256, 0, stream>>>(
      mb, wfc1T, fc1_b, h1);

  // FC2 + residual -> out
  k_gemm_fc2<<<(TOKENS / 16) * (CC / 64) / 8, 256, 0, stream>>>(
      h1, wfc2T, fc2_b, yb, (float*)d_out);
}
